// BasicDeconvolutionBlock_32143535243817
// MI455X (gfx1250) — matrix operand order repaired and checked by reading, not compiled, not run on hardware
//
#include <hip/hip_runtime.h>
#include <hip/hip_bf16.h>
#include <stdint.h>

#define NTOT   200000
#define INC    64
#define OUTC   64
#define NK     27
#define EPSV   1e-5f
#define SLOPE  0.01f

typedef __attribute__((ext_vector_type(16))) __bf16 v16bf;
typedef __attribute__((ext_vector_type(8)))  float  v8f;

union ABfrag { uint4 q[2]; v16bf v; };

__device__ __forceinline__ uint16_t f2bf(float f) {
    uint32_t u = __float_as_uint(f);
    uint32_t r = u + 0x7FFFu + ((u >> 16) & 1u);   // round to nearest even
    return (uint16_t)(r >> 16);
}

// ---------------------------------------------------------------- phase 0a: x -> bf16
__global__ __launch_bounds__(256) void cvt_x_kernel(const float* __restrict__ x,
                                                    uint32_t* __restrict__ xb) {
    size_t i = (size_t)blockIdx.x * 256 + threadIdx.x;      // 6.4M packed pairs
    float2 v = ((const float2*)x)[i];
    xb[i] = (uint32_t)f2bf(v.x) | ((uint32_t)f2bf(v.y) << 16);
}

// ---------------------------------------------------------------- phase 0b: W[k][ci][co] -> Wt[k][co][ci] bf16
__global__ __launch_bounds__(256) void cvt_w_kernel(const float* __restrict__ W,
                                                    uint16_t* __restrict__ wt) {
    int i = blockIdx.x * 256 + threadIdx.x;                 // 27*4096 = 110592 exact
    int k  = i >> 12;
    int r  = i & 4095;
    int ci = r >> 6;
    int co = r & 63;
    wt[(k << 12) + (co << 6) + ci] = f2bf(W[i]);
}

// ---------------------------------------------------------------- phase 1: gather-GEMM-accumulate (WMMA)
__device__ __forceinline__ void async_copy_w(const uint16_t* src, const uint16_t* dstLds, int tid) {
    uint32_t ldsa = (uint32_t)(uintptr_t)dstLds + (uint32_t)tid * 16u;
    uint64_t ga   = (uint64_t)(uintptr_t)src + (uint64_t)tid * 16u;
    asm volatile("global_load_async_to_lds_b128 %0, %1, off"             :: "v"(ldsa), "v"(ga) : "memory");
    asm volatile("global_load_async_to_lds_b128 %0, %1, off offset:2048" :: "v"(ldsa), "v"(ga) : "memory");
    asm volatile("global_load_async_to_lds_b128 %0, %1, off offset:4096" :: "v"(ldsa), "v"(ga) : "memory");
    asm volatile("global_load_async_to_lds_b128 %0, %1, off offset:6144" :: "v"(ldsa), "v"(ga) : "memory");
}

__device__ __forceinline__ void wait_async0() {
    asm volatile("s_wait_asynccnt 0x0" ::: "memory");
}

__global__ __launch_bounds__(128) void conv_wmma_kernel(const uint16_t* __restrict__ xb,
                                                        const uint16_t* __restrict__ wt,
                                                        const int* __restrict__ nbr,
                                                        float* __restrict__ out) {
    __shared__ uint16_t smW[2][64 * 64];                    // 2 x 8KB double buffer

    const int tid  = threadIdx.x;
    const int lane = tid & 31;
    const int wid  = tid >> 5;
    const int half = (lane >> 4) & 1;                       // lane group within wave
    const int l15  = lane & 15;
    const int mbase = blockIdx.x * 64 + wid * 16;           // 3125 blocks * 64 rows = 200000 exact
    const int mrow  = mbase + l15;

    v8f acc[4];
    #pragma unroll
    for (int nt = 0; nt < 4; ++nt) acc[nt] = (v8f){0.f,0.f,0.f,0.f,0.f,0.f,0.f,0.f};

    // prefetch Wt[0]
    async_copy_w(wt, &smW[0][0], tid);
    wait_async0();
    __syncthreads();

    for (int k = 0; k < NK; ++k) {
        const int cur = k & 1;
        if (k + 1 < NK)
            async_copy_w(wt + (size_t)(k + 1) * 4096, &smW[cur ^ 1][0], tid);

        const int idx = nbr[k * NTOT + mrow];
        const uint16_t* bufp = &smW[cur][0];

        #pragma unroll
        for (int kt = 0; kt < 2; ++kt) {
            // ---- A fragment: 16x32 bf16, M=voxels, K=ci (kt*32 .. kt*32+31)
            ABfrag a;
            a.q[0] = make_uint4(0u, 0u, 0u, 0u);
            a.q[1] = make_uint4(0u, 0u, 0u, 0u);
            if (idx >= 0) {
                const uint16_t* base = xb + (size_t)idx * INC + kt * 32 + half * 8;
                a.q[0] = *(const uint4*)(base);        // K = cib .. cib+7
                a.q[1] = *(const uint4*)(base + 16);   // K = cib+16 .. cib+23
            }
            // ---- 4 B fragments (co tiles) + WMMA
            #pragma unroll
            for (int nt = 0; nt < 4; ++nt) {
                const uint16_t* brow = bufp + (size_t)(nt * 16 + l15) * 64 + kt * 32 + half * 8;
                ABfrag b;
                b.q[0] = *(const uint4*)(brow);
                b.q[1] = *(const uint4*)(brow + 16);
                acc[nt] = __builtin_amdgcn_wmma_f32_16x16x32_bf16(
                    /*neg_a=*/false, a.v, /*neg_b=*/false, b.v,
                    /*c_mod=*/(short)0, acc[nt], /*reuse_a=*/false, /*reuse_b=*/false);
            }
        }
        wait_async0();
        __syncthreads();
    }

    // ---- store h tile: C/D layout: lane%16 -> column, vgpr r -> row (+8 for lanes>=16)
    const int m0 = mbase + half * 8;
    #pragma unroll
    for (int nt = 0; nt < 4; ++nt) {
        const int co = nt * 16 + l15;
        #pragma unroll
        for (int r = 0; r < 8; ++r)
            out[(size_t)(m0 + r) * OUTC + co] = acc[nt][r];
    }
}

// ---------------------------------------------------------------- phase 2a: per-block channel partials (deterministic, coalesced)
__global__ __launch_bounds__(256) void stats_part_kernel(const float* __restrict__ h,
                                                         float* __restrict__ part) {
    const int b = blockIdx.x;                 // 500 blocks * 400 rows = 200000 exact
    const int c = threadIdx.x & 63;
    const int g = threadIdx.x >> 6;           // 0..3
    const size_t base = (size_t)b * 400;
    float s = 0.f, s2 = 0.f;
    for (int r = g; r < 400; r += 4) {
        float v = h[(base + r) * OUTC + c];
        s += v; s2 += v * v;
    }
    __shared__ float sh[256], sh2[256];
    sh[threadIdx.x] = s; sh2[threadIdx.x] = s2;
    __syncthreads();
    if (g == 0) {
        float ts = sh[c]  + sh[64 + c]  + sh[128 + c]  + sh[192 + c];
        float t2 = sh2[c] + sh2[64 + c] + sh2[128 + c] + sh2[192 + c];
        part[b * 128 + c]      = ts;
        part[b * 128 + 64 + c] = t2;
    }
}

// ---------------------------------------------------------------- phase 2b: finalize mean/var -> scale/shift
__global__ __launch_bounds__(64) void stats_final_kernel(const float* __restrict__ part,
                                                         const float* __restrict__ gamma,
                                                         const float* __restrict__ beta,
                                                         float* __restrict__ ss) {
    const int c = threadIdx.x;
    float s = 0.f, s2 = 0.f;
    for (int b = 0; b < 500; ++b) {
        s  += part[b * 128 + c];
        s2 += part[b * 128 + 64 + c];
    }
    const float invn = 1.0f / (float)NTOT;
    float mean = s * invn;
    float var  = s2 * invn - mean * mean;
    float sc   = gamma[c] * rsqrtf(var + EPSV);
    ss[c]      = sc;
    ss[64 + c] = beta[c] - mean * sc;
}

// ---------------------------------------------------------------- phase 3: normalize + LeakyReLU (in place, float4)
__global__ __launch_bounds__(256) void norm_kernel(float* __restrict__ out,
                                                   const float* __restrict__ ss) {
    __shared__ float s_sc[64], s_sh[64];
    if (threadIdx.x < 64) {
        s_sc[threadIdx.x] = ss[threadIdx.x];
        s_sh[threadIdx.x] = ss[64 + threadIdx.x];
    }
    __syncthreads();
    size_t i = (size_t)blockIdx.x * 256 + threadIdx.x;      // 3.2M float4
    float4* p = (float4*)out + i;
    float4 v = *p;
    int c0 = (int)((i * 4) & 63);
    float a, b;
    a = v.x * s_sc[c0]     + s_sh[c0];     v.x = a >= 0.f ? a : SLOPE * a;
    b = v.y * s_sc[c0 + 1] + s_sh[c0 + 1]; v.y = b >= 0.f ? b : SLOPE * b;
    a = v.z * s_sc[c0 + 2] + s_sh[c0 + 2]; v.z = a >= 0.f ? a : SLOPE * a;
    b = v.w * s_sc[c0 + 3] + s_sh[c0 + 3]; v.w = b >= 0.f ? b : SLOPE * b;
    *p = v;
}

// ---------------------------------------------------------------- launcher
extern "C" void kernel_launch(void* const* d_in, const int* in_sizes, int n_in,
                              void* d_out, int out_size, void* d_ws, size_t ws_size,
                              hipStream_t stream) {
    (void)in_sizes; (void)n_in; (void)out_size; (void)ws_size;

    const float* x     = (const float*)d_in[0];   // [N, 64]
    const float* W     = (const float*)d_in[1];   // [27, 64, 64]
    const float* gamma = (const float*)d_in[2];   // [64]
    const float* beta  = (const float*)d_in[3];   // [64]
    const int*   nbr   = (const int*)d_in[4];     // [27, N]
    float* out = (float*)d_out;                   // [N, 64]

    // workspace layout
    uint16_t* xb  = (uint16_t*)d_ws;                        // N*64 bf16      (25.6 MB)
    uint16_t* wtp = xb + (size_t)NTOT * INC;                // 27*4096 bf16   (221 KB)
    float*    ss  = (float*)(wtp + (size_t)NK * 4096);      // 128 floats
    float*    prt = ss + 128;                               // 500*128 floats (256 KB)

    // phase 0: precision/layout staging
    cvt_x_kernel<<<(NTOT * INC / 2) / 256, 256, 0, stream>>>(x, (uint32_t*)xb);
    cvt_w_kernel<<<(NK * 4096) / 256, 256, 0, stream>>>(W, wtp);

    // phase 1: 27x gather-GEMM accumulate via bf16 WMMA, h -> d_out
    conv_wmma_kernel<<<NTOT / 64, 128, 0, stream>>>(xb, wtp, nbr, out);

    // phase 2: deterministic batch statistics -> scale/shift
    stats_part_kernel<<<500, 256, 0, stream>>>(out, prt);
    stats_final_kernel<<<1, 64, 0, stream>>>(prt, gamma, beta, ss);

    // phase 3: BN apply + LeakyReLU in place
    norm_kernel<<<(NTOT * OUTC / 4) / 256, 256, 0, stream>>>(out, ss);
}
